// RNNDecoder_2465311228486
// MI455X (gfx1250) — matrix operand order repaired and checked by reading, not compiled, not run on hardware
//
#include <hip/hip_runtime.h>
#include <hip/hip_bf16.h>
#include <math.h>

// ---------------------------------------------------------------------------
// GRU decoder + attention + 2-layer head for MI455X / gfx1250.
// All GEMMs run on v_wmma_f32_16x16x32_f16 with f16 operands and fp32
// accumulate. Weights are converted to f16 once per call (fc2: 65 MB,
// L2-resident across all 64 steps); activations flow in f16 between GEMMs.
// The GEMM inner loop is software-pipelined (prefetch next K-slab into a
// second register set) so L2 load latency overlaps the 4 WMMAs per slab —
// critical because fc2 only spawns ~1000 waves (~1/SIMD). Logits use
// non-temporal stores so the 262 MB output stream never evicts the
// L2-resident weights.
// ---------------------------------------------------------------------------

typedef __attribute__((ext_vector_type(16))) _Float16 v16h;
typedef __attribute__((ext_vector_type(8)))  _Float16 v8h;
typedef __attribute__((ext_vector_type(8)))  float    v8f;

#define VSZ 32000
#define ESZ 256
#define HSZ 512
#define FCZ 1024
#define BSZ 32
#define NSZ 128
#define TSZ 64

// ---------------------------------------------------------------------------
// helpers
// ---------------------------------------------------------------------------
__global__ void k_f32_to_f16(const float* __restrict__ s, _Float16* __restrict__ d, int n) {
    int i = blockIdx.x * blockDim.x + threadIdx.x;
    if (i < n) d[i] = (_Float16)s[i];
}

__global__ void k_zero_h(_Float16* __restrict__ p, int n) {
    int i = blockIdx.x * blockDim.x + threadIdx.x;
    if (i < n) p[i] = (_Float16)0.0f;
}

// rnn_in[b,:] = concat(emb[token[b,t]], ctx[b,:]) -> (32, 768) f16
__global__ void k_build_rnn_in(const float* __restrict__ emb,
                               const int*   __restrict__ tokens,
                               const _Float16* __restrict__ ctx,
                               _Float16* __restrict__ rnn_in, int t) {
    int i = blockIdx.x * blockDim.x + threadIdx.x;   // 32*768
    if (i >= BSZ * (ESZ + HSZ)) return;
    int b = i / (ESZ + HSZ);
    int c = i % (ESZ + HSZ);
    _Float16 v;
    if (c < ESZ) {
        int tok = tokens[b * TSZ + t];
        v = (_Float16)emb[(size_t)tok * ESZ + c];
    } else {
        v = ctx[b * HSZ + (c - ESZ)];
    }
    rnn_in[i] = v;
}

// GRU gates in exact fp32; emits h_new in f32 (attention + next-step gates)
// and f16 (next-step gh GEMM operand)
__global__ void k_gru_gates(const float* __restrict__ gi,
                            const float* __restrict__ gh,
                            const float* __restrict__ h,
                            float* __restrict__ h_new,
                            _Float16* __restrict__ h_new_h) {
    int i = blockIdx.x * blockDim.x + threadIdx.x;   // 32*512
    if (i >= BSZ * HSZ) return;
    int b = i / HSZ, j = i % HSZ;
    const float* gib = gi + b * 3 * HSZ;
    const float* ghb = gh + b * 3 * HSZ;
    float r = 1.0f / (1.0f + expf(-(gib[j]       + ghb[j])));
    float z = 1.0f / (1.0f + expf(-(gib[HSZ + j] + ghb[HSZ + j])));
    float n = tanhf(gib[2 * HSZ + j] + r * ghb[2 * HSZ + j]);
    float hv = (1.0f - z) * n + z * h[i];
    h_new[i]   = hv;
    h_new_h[i] = (_Float16)hv;
}

// dot-product attention; outputs ctx (f16, feeds rnn_in) and pred=[ctx,h] (f16, feeds fc1)
__global__ void k_attention(const float* __restrict__ enc,     // (B,N,H) f32
                            const float* __restrict__ h_new,   // (B,H) f32
                            _Float16* __restrict__ ctx_h,      // (B,H) f16
                            _Float16* __restrict__ pred_h) {   // (B,2H) f16
    __shared__ float sh[HSZ];
    __shared__ float sp[NSZ];
    __shared__ float sinv;
    int b = blockIdx.x;
    int tid = threadIdx.x;
    const float* encb = enc + (size_t)b * NSZ * HSZ;
    for (int j = tid; j < HSZ; j += 256) sh[j] = h_new[b * HSZ + j];
    __syncthreads();
    if (tid < NSZ) {
        const float* er = encb + tid * HSZ;
        float s = 0.0f;
        for (int k = 0; k < HSZ; ++k) s += er[k] * sh[k];
        sp[tid] = s;
    }
    __syncthreads();
    if (tid == 0) {
        float m = sp[0];
        for (int n = 1; n < NSZ; ++n) m = fmaxf(m, sp[n]);
        float sum = 0.0f;
        for (int n = 0; n < NSZ; ++n) { float e = expf(sp[n] - m); sp[n] = e; sum += e; }
        sinv = 1.0f / sum;
    }
    __syncthreads();
    float inv = sinv;
    for (int j = tid; j < HSZ; j += 256) {
        float acc = 0.0f;
        for (int n = 0; n < NSZ; ++n) acc += sp[n] * encb[n * HSZ + j];
        float c = acc * inv;
        ctx_h[b * HSZ + j]            = (_Float16)c;
        pred_h[b * 2 * HSZ + j]       = (_Float16)c;
        pred_h[b * 2 * HSZ + HSZ + j] = (_Float16)sh[j];
    }
}

// ---------------------------------------------------------------------------
// WMMA GEMM:  C(32, N) = A(32, K) @ W(N, K)^T + bias   (K%32==0, N%64==0)
// A, W row-major f16. Outputs: CF (f32, optional) and/or CH (f16, optional).
// act=1 -> ELU; nt=1 -> non-temporal f32 stores.
// Wave computes 16(M) x 64(N); inner loop prefetches next K-slab (2-stage SWP).
// ---------------------------------------------------------------------------
__device__ inline v16h load_a_frag(const _Float16* __restrict__ p) {
    // A 16x32 f16, contiguous k order used here (B matches): lane (row = lane&15) holds K = {0..15} (+16 upper half)
    v8h lo = *(const v8h*)(p);
    v8h hi = *(const v8h*)(p + 8);
    return __builtin_shufflevector(lo, hi, 0, 1, 2, 3, 4, 5, 6, 7,
                                           8, 9, 10, 11, 12, 13, 14, 15);
}

__device__ inline void store_tile(v8f c, float bs,
                                  float* __restrict__ CF, _Float16* __restrict__ CH,
                                  int rowBase, int ldC, int col, int act, int nt) {
#pragma unroll
    for (int r = 0; r < 8; ++r) {
        float x = c[r] + bs;
        if (act) x = (x > 0.0f) ? x : (expf(x) - 1.0f);
        size_t off = (size_t)(rowBase + r) * ldC + col;
        if (CF) {
            if (nt) __builtin_nontemporal_store(x, CF + off);
            else    CF[off] = x;
        }
        if (CH) CH[off] = (_Float16)x;
    }
}

__global__ __launch_bounds__(256)
void k_gemm_wmma(const _Float16* __restrict__ A, const _Float16* __restrict__ W,
                 const float* __restrict__ bias,
                 float* __restrict__ CF, _Float16* __restrict__ CH,
                 int K, int N, int ldC, int act, int nt) {
    const int lane = threadIdx.x & 31;
    const int wave = threadIdx.x >> 5;
    const int gw   = blockIdx.x * (blockDim.x >> 5) + wave;
    const int mt   = gw & 1;          // M = 32 -> 2 row tiles
    const int ng   = gw >> 1;         // group of 4 N-tiles (64 cols)
    const int n0   = ng * 64;
    if (n0 >= N) return;              // wave-uniform guard (EXEC stays full)

    const int khalfA  = (lane & 16) ? 16 : 0;   // A 16x32 f16 layout
    const int khalfB  = (lane & 16) ? 16 : 0;   // B 32x16 f16 layout
    const int colBase = lane & 15;
    const int row     = mt * 16 + (lane & 15);

    const _Float16* arow = A + (size_t)row * K + khalfA;
    const _Float16* w0 = W + (size_t)(n0 +  0 + colBase) * K + khalfB;
    const _Float16* w1 = W + (size_t)(n0 + 16 + colBase) * K + khalfB;
    const _Float16* w2 = W + (size_t)(n0 + 32 + colBase) * K + khalfB;
    const _Float16* w3 = W + (size_t)(n0 + 48 + colBase) * K + khalfB;

    v8f c0 = {}, c1 = {}, c2 = {}, c3 = {};

    // ---- software pipeline: stage 0 ----
    v16h a  = load_a_frag(arow);
    v16h b0 = *(const v16h*)(w0);
    v16h b1 = *(const v16h*)(w1);
    v16h b2 = *(const v16h*)(w2);
    v16h b3 = *(const v16h*)(w3);

    for (int k0 = 32; k0 < K; k0 += 32) {
        // prefetch next slab into a second register set (loads issue before
        // the WMMAs below; waits land at the top of the next iteration)
        v16h an  = load_a_frag(arow + k0);
        v16h nb0 = *(const v16h*)(w0 + k0);
        v16h nb1 = *(const v16h*)(w1 + k0);
        v16h nb2 = *(const v16h*)(w2 + k0);
        v16h nb3 = *(const v16h*)(w3 + k0);

        c0 = __builtin_amdgcn_wmma_f32_16x16x32_f16(false, a, false, b0, (short)0, c0, false, false);
        c1 = __builtin_amdgcn_wmma_f32_16x16x32_f16(false, a, false, b1, (short)0, c1, false, false);
        c2 = __builtin_amdgcn_wmma_f32_16x16x32_f16(false, a, false, b2, (short)0, c2, false, false);
        c3 = __builtin_amdgcn_wmma_f32_16x16x32_f16(false, a, false, b3, (short)0, c3, false, false);

        a = an; b0 = nb0; b1 = nb1; b2 = nb2; b3 = nb3;
    }
    // ---- epilogue ----
    c0 = __builtin_amdgcn_wmma_f32_16x16x32_f16(false, a, false, b0, (short)0, c0, false, false);
    c1 = __builtin_amdgcn_wmma_f32_16x16x32_f16(false, a, false, b1, (short)0, c1, false, false);
    c2 = __builtin_amdgcn_wmma_f32_16x16x32_f16(false, a, false, b2, (short)0, c2, false, false);
    c3 = __builtin_amdgcn_wmma_f32_16x16x32_f16(false, a, false, b3, (short)0, c3, false, false);

    // D layout: VGPR r -> row mt*16 + r + (lane>=16 ? 8 : 0); col = n0+sub*16+(lane&15)
    const int rowBase = mt * 16 + ((lane & 16) ? 8 : 0);
    store_tile(c0, bias[n0 +  0 + colBase], CF, CH, rowBase, ldC, n0 +  0 + colBase, act, nt);
    store_tile(c1, bias[n0 + 16 + colBase], CF, CH, rowBase, ldC, n0 + 16 + colBase, act, nt);
    store_tile(c2, bias[n0 + 32 + colBase], CF, CH, rowBase, ldC, n0 + 32 + colBase, act, nt);
    store_tile(c3, bias[n0 + 48 + colBase], CF, CH, rowBase, ldC, n0 + 48 + colBase, act, nt);
}

// ---------------------------------------------------------------------------
// host-side orchestration (graph-capture safe: only kernel launches on stream)
// ---------------------------------------------------------------------------
extern "C" void kernel_launch(void* const* d_in, const int* in_sizes, int n_in,
                              void* d_out, int out_size, void* d_ws, size_t ws_size,
                              hipStream_t stream) {
    (void)in_sizes; (void)n_in; (void)out_size; (void)ws_size;

    const float* enc    = (const float*)d_in[0];   // (B,N,H)
    const float* h_init = (const float*)d_in[1];   // (B,H)
    const int*   tokens = (const int*)  d_in[2];   // (B,T)
    const float* emb    = (const float*)d_in[3];   // (V,E)
    const float* W_ih   = (const float*)d_in[4];   // (3H, E+H)
    const float* W_hh   = (const float*)d_in[5];   // (3H, H)
    const float* b_ih   = (const float*)d_in[6];
    const float* b_hh   = (const float*)d_in[7];
    const float* W_map  = (const float*)d_in[8];   // (H,H)
    const float* b_map  = (const float*)d_in[9];
    const float* W_fc1  = (const float*)d_in[10];  // (FC, 2H)
    const float* b_fc1  = (const float*)d_in[11];
    const float* W_fc2  = (const float*)d_in[12];  // (V, FC)
    const float* b_fc2  = (const float*)d_in[13];
    float* out = (float*)d_out;                    // (B,T,V)

    // ---- workspace layout: f16 region first (all sizes multiples of 16) ----
    _Float16* wsh = (_Float16*)d_ws;
    size_t o = 0;
    auto alloc_h = [&](size_t n) { _Float16* p = wsh + o; o += n; return p; };
    _Float16* Wih_h   = alloc_h((size_t)3 * HSZ * (ESZ + HSZ));
    _Float16* Whh_h   = alloc_h((size_t)3 * HSZ * HSZ);
    _Float16* Wmap_h  = alloc_h((size_t)HSZ * HSZ);
    _Float16* Wfc1_h  = alloc_h((size_t)FCZ * 2 * HSZ);
    _Float16* Wfc2_h  = alloc_h((size_t)VSZ * FCZ);
    _Float16* hinit_h = alloc_h(BSZ * HSZ);
    _Float16* rnnin_h = alloc_h(BSZ * (ESZ + HSZ));
    _Float16* ctx_h[2] = { alloc_h(BSZ * HSZ), alloc_h(BSZ * HSZ) };
    _Float16* h_h[2]   = { alloc_h(BSZ * HSZ), alloc_h(BSZ * HSZ) };
    _Float16* pred_h  = alloc_h(BSZ * 2 * HSZ);
    _Float16* hid_h   = alloc_h(BSZ * FCZ);
    // f32 region
    float* wsf = (float*)(wsh + o);
    float* hbuf[2] = { wsf, wsf + BSZ * HSZ };
    float* gi = wsf + 2 * BSZ * HSZ;       // 32*1536
    float* gh = gi + BSZ * 3 * HSZ;        // 32*1536

    auto conv = [&](const float* s, _Float16* d, size_t n) {
        k_f32_to_f16<<<(unsigned)((n + 255) / 256), 256, 0, stream>>>(s, d, (int)n);
    };
    auto gemm = [&](const _Float16* A, const _Float16* W, const float* bias,
                    float* CF, _Float16* CH, int K, int N, int ldC, int act, int nt) {
        int waves  = 2 * (N / 64);
        int blocks = (waves + 7) / 8;      // 8 waves (256 threads) per block
        k_gemm_wmma<<<blocks, 256, 0, stream>>>(A, W, bias, CF, CH, K, N, ldC, act, nt);
    };

    // ---- one-time (per call) weight / init conversions ----
    conv(W_ih,  Wih_h,  (size_t)3 * HSZ * (ESZ + HSZ));
    conv(W_hh,  Whh_h,  (size_t)3 * HSZ * HSZ);
    conv(W_map, Wmap_h, (size_t)HSZ * HSZ);
    conv(W_fc1, Wfc1_h, (size_t)FCZ * 2 * HSZ);
    conv(W_fc2, Wfc2_h, (size_t)VSZ * FCZ);
    conv(h_init, hinit_h, BSZ * HSZ);
    k_zero_h<<<(BSZ * HSZ + 255) / 256, 256, 0, stream>>>(ctx_h[0], BSZ * HSZ);

    // h0 = h_init @ W_map^T + b_map  (f32 for gate math, f16 for gh GEMM)
    gemm(hinit_h, Wmap_h, b_map, hbuf[0], h_h[0], HSZ, HSZ, HSZ, 0, 0);

    for (int t = 0; t < TSZ; ++t) {
        float*    hc  = hbuf[t & 1];
        float*    hn  = hbuf[(t + 1) & 1];
        _Float16* hch = h_h[t & 1];
        _Float16* hnh = h_h[(t + 1) & 1];
        _Float16* cch = ctx_h[t & 1];
        _Float16* cnh = ctx_h[(t + 1) & 1];

        k_build_rnn_in<<<(BSZ * (ESZ + HSZ) + 255) / 256, 256, 0, stream>>>(
            emb, tokens, cch, rnnin_h, t);
        gemm(rnnin_h, Wih_h, b_ih, gi, nullptr, ESZ + HSZ, 3 * HSZ, 3 * HSZ, 0, 0);
        gemm(hch,     Whh_h, b_hh, gh, nullptr, HSZ,       3 * HSZ, 3 * HSZ, 0, 0);
        k_gru_gates<<<(BSZ * HSZ + 255) / 256, 256, 0, stream>>>(gi, gh, hc, hn, hnh);
        k_attention<<<BSZ, 256, 0, stream>>>(enc, hn, cnh, pred_h);
        gemm(pred_h, Wfc1_h, b_fc1, nullptr, hid_h, 2 * HSZ, FCZ, FCZ, 1, 0);
        // logits for step t -> d_out slice; ldC = T*V; non-temporal stores
        gemm(hid_h, Wfc2_h, b_fc2, out + (size_t)t * VSZ, nullptr, FCZ, VSZ, TSZ * VSZ, 0, 1);
    }
}
